// GramDistMap_47425028882917
// MI455X (gfx1250) — matrix operand order repaired and checked by reading, not compiled, not run on hardware
//
#include <hip/hip_runtime.h>

// GramDistMap, parallel path. B=8, C=8, N=1024, D=256.
// out[b,n,m] = mean_c (sq[n]+sq[m]-2*G.G^T) / (norm[n]*norm[m])
//
// Roofline: 64 batched 1024x1024x256 GEMMs = 34 GFLOP vs ~100MB traffic -> compute bound.
// ISA: V_WMMA_F32_16X16X32_BF16 does 8x the MACs/instruction of V_WMMA_F32_16X16X4_F32
// and co-executes with VALU (16-bit XDL ops tracked as TRANS). So: bf16x3 split GEMM
// (hi*hi + hi*lo + lo*hi, fp32 accumulate) = fp32-class accuracy at ~2.7x fewer, faster
// matrix instructions. Symmetric 128x128 tile pruning removes a further 44% of compute.

typedef __attribute__((ext_vector_type(8)))  __bf16 v8bf;
typedef __attribute__((ext_vector_type(16))) __bf16 v16bf;
typedef __attribute__((ext_vector_type(8)))  float  v8f;

#define GD_B 8
#define GD_C 8
#define GD_N 1024
#define GD_D 256
#define LDSB 40   // bf16 LDS row stride: 32 + 8 pad (80B rows: 16B-aligned, full bank spread)

// round-to-nearest-even fp32 -> bf16 (bits)
static __device__ __forceinline__ unsigned int f2bf_rn(float x) {
    unsigned int u = __float_as_uint(x);
    return (u + 0x7FFFu + ((u >> 16) & 1u)) >> 16;
}
// split two floats into packed hi/lo bf16 words (hi+lo captures ~17 mantissa bits)
static __device__ __forceinline__ void split2(float x, float y,
                                              unsigned int& ph, unsigned int& pl) {
    unsigned int hx = f2bf_rn(x), hy = f2bf_rn(y);
    float rx = x - __uint_as_float(hx << 16);   // exact (Sterbenz-range subtraction)
    float ry = y - __uint_as_float(hy << 16);
    ph = hx | (hy << 16);
    pl = f2bf_rn(rx) | (f2bf_rn(ry) << 16);
}
// build a v16bf fragment from two 16-byte LDS chunks
static __device__ __forceinline__ v16bf ldfrag(const unsigned short* p0,
                                               const unsigned short* p1) {
    v8bf a = *(const v8bf*)p0;
    v8bf b = *(const v8bf*)p1;
    return __builtin_shufflevector(a, b, 0,1,2,3,4,5,6,7,8,9,10,11,12,13,14,15);
}

// ---------------- Kernel 1: per-row sq = sum(g^2), norm = sqrt(sum(g^4)), fp32 ----------
__global__ __launch_bounds__(256)
void gram_rownorm_kernel(const float* __restrict__ g,
                         float* __restrict__ sqArr, float* __restrict__ nrArr) {
    const int t    = threadIdx.x;
    const int lane = t & 31, wave = t >> 5;
    const int sub = lane & 7, subrow = lane >> 3;        // 8 lanes per row
    const int row = blockIdx.x * 32 + wave * 4 + subrow;
    const float4* gp = (const float4*)g + (long)row * (GD_D / 4);
    float s2 = 0.f, s4 = 0.f;
#pragma unroll
    for (int i = 0; i < 8; ++i) {
        float4 v = gp[i * 8 + sub];
        float p;
        p = v.x * v.x; s2 += p; s4 += p * p;
        p = v.y * v.y; s2 += p; s4 += p * p;
        p = v.z * v.z; s2 += p; s4 += p * p;
        p = v.w * v.w; s2 += p; s4 += p * p;
    }
#pragma unroll
    for (int d = 1; d < 8; d <<= 1) {
        s2 += __shfl_xor(s2, d, 32);
        s4 += __shfl_xor(s4, d, 32);
    }
    if (sub == 0) { sqArr[row] = s2; nrArr[row] = sqrtf(s4); }
}

// ---------------- Kernel 2: bf16x3 symmetric tiled GEMM + normalize + mean over C -------
// Grid: x = 36 upper-triangular (ti,tj) 128x128 tile pairs, z = b. 256 threads = 8 waves.
// Wave tile 32x64 = 2x4 WMMA tiles; K loop: c=0..7, k-chunks of 32 (one WMMA K-step).
__global__ __launch_bounds__(256)
void gram_gemm_kernel(const float* __restrict__ g,
                      const float* __restrict__ sqArr, const float* __restrict__ nrArr,
                      float* __restrict__ out) {
    __shared__ __align__(16) unsigned short lds_ah[128 * LDSB];
    __shared__ __align__(16) unsigned short lds_al[128 * LDSB];
    __shared__ __align__(16) unsigned short lds_bh[128 * LDSB];
    __shared__ __align__(16) unsigned short lds_bl[128 * LDSB];
    __shared__ float s_sqM[128], s_nrM[128], s_sqN[128], s_nrN[128];

    const int t    = threadIdx.x;
    const int wave = t >> 5, lane = t & 31;
    const int lo = lane & 15, hi = lane >> 4;
    const int wm = wave & 3, wn = wave >> 2;
    const int moff = wm * 32, noff = wn * 64;

    // decode upper-triangular tile pair (ti <= tj); uniform across the wave
    int pid = blockIdx.x;
    int ti = 0;
    while (pid >= (8 - ti)) { pid -= (8 - ti); ++ti; }
    const int tj = ti + pid;
    const int b  = blockIdx.z;

    const v8f vzero = {};
    v8f acc_out[2][4];
    v8f acc_dot[2][4];
#pragma unroll
    for (int mi = 0; mi < 2; ++mi)
#pragma unroll
        for (int ni = 0; ni < 4; ++ni) acc_out[mi][ni] = vzero;

    for (int c = 0; c < GD_C; ++c) {
        const long rowBase = ((long)b * GD_C + c) * GD_N;
        const float* gA = g + (rowBase + (long)ti * 128) * GD_D;
        const float* gB = g + (rowBase + (long)tj * 128) * GD_D;

        __syncthreads();  // previous-c epilogue done reading s_* before overwrite
        if (t < 128) {
            s_sqM[t] = sqArr[rowBase + ti * 128 + t];
            s_nrM[t] = nrArr[rowBase + ti * 128 + t];
        } else {
            const int tt = t - 128;
            s_sqN[tt] = sqArr[rowBase + tj * 128 + tt];
            s_nrN[tt] = nrArr[rowBase + tj * 128 + tt];
        }

#pragma unroll
        for (int mi = 0; mi < 2; ++mi)
#pragma unroll
            for (int ni = 0; ni < 4; ++ni) acc_dot[mi][ni] = vzero;

        for (int k0 = 0; k0 < GD_D; k0 += 32) {
            __syncthreads();  // WAR on LDS tiles (+ s_* visibility on first chunk)
            // load 128x32 fp32 per matrix, split to hi/lo bf16 tiles (4 float4/thread/tile)
#pragma unroll
            for (int it = 0; it < 4; ++it) {
                const int idx = t + it * 256;
                const int r = idx >> 3, c4 = (idx & 7) << 2;
                const float4 va = *(const float4*)(gA + (long)r * GD_D + k0 + c4);
                const float4 vb = *(const float4*)(gB + (long)r * GD_D + k0 + c4);
                uint2 ah, al, bh, bl;
                split2(va.x, va.y, ah.x, al.x);
                split2(va.z, va.w, ah.y, al.y);
                split2(vb.x, vb.y, bh.x, bl.x);
                split2(vb.z, vb.w, bh.y, bl.y);
                const int off = r * LDSB + c4;
                *(uint2*)&lds_ah[off] = ah;
                *(uint2*)&lds_al[off] = al;
                *(uint2*)&lds_bh[off] = bh;
                *(uint2*)&lds_bl[off] = bl;
            }
            __syncthreads();

            // one 16x16x32 K-step: A lane(M=lo): K chunks at hi*8 and 16+hi*8;
            //                      B lane(N=lo): K chunks at hi*8 and 16+hi*8.
            v16bf ah[2], al2[2];
#pragma unroll
            for (int mi = 0; mi < 2; ++mi) {
                const int ro = (moff + mi * 16 + lo) * LDSB + hi * 8;
                ah[mi]  = ldfrag(&lds_ah[ro], &lds_ah[ro + 16]);
                al2[mi] = ldfrag(&lds_al[ro], &lds_al[ro + 16]);
            }
#pragma unroll
            for (int ni = 0; ni < 4; ++ni) {
                const int ro = (noff + ni * 16 + lo) * LDSB + hi * 8;
                const v16bf bh = ldfrag(&lds_bh[ro], &lds_bh[ro + 16]);
                const v16bf bl = ldfrag(&lds_bl[ro], &lds_bl[ro + 16]);
#pragma unroll
                for (int mi = 0; mi < 2; ++mi) {
                    v8f acc = acc_dot[mi][ni];
                    acc = __builtin_amdgcn_wmma_f32_16x16x32_bf16(
                        false, ah[mi],  false, bh, (short)0, acc, false, false);
                    acc = __builtin_amdgcn_wmma_f32_16x16x32_bf16(
                        false, ah[mi],  false, bl, (short)0, acc, false, false);
                    acc = __builtin_amdgcn_wmma_f32_16x16x32_bf16(
                        false, al2[mi], false, bh, (short)0, acc, false, false);
                    acc_dot[mi][ni] = acc;
                }
            }
        }

        // per-c epilogue: fold (sq_m + sq_n - 2*dot) / (norm_m * norm_n) into acc_out
#pragma unroll
        for (int mi = 0; mi < 2; ++mi) {
            const int mL = moff + mi * 16 + 8 * hi;
#pragma unroll
            for (int ni = 0; ni < 4; ++ni) {
                const int nL = noff + ni * 16 + lo;
                const float sqn = s_sqN[nL], nrn = s_nrN[nL];
#pragma unroll
                for (int v = 0; v < 8; ++v) {
                    const float sqm = s_sqM[mL + v];
                    const float nrm = s_nrM[mL + v];
                    acc_out[mi][ni][v] +=
                        (sqm + sqn - 2.0f * acc_dot[mi][ni][v]) / (nrm * nrn);
                }
            }
        }
    }

    // write tile (and mirror for off-diagonal pairs); mean over C = *1/8
    const long ob = (long)b * GD_N * GD_N;
#pragma unroll
    for (int mi = 0; mi < 2; ++mi) {
#pragma unroll
        for (int ni = 0; ni < 4; ++ni) {
            const int mg = ti * 128 + moff + mi * 16 + 8 * hi;
            const int ng = tj * 128 + noff + ni * 16 + lo;
#pragma unroll
            for (int v = 0; v < 8; ++v) {
                const float val = acc_out[mi][ni][v] * 0.125f;
                out[ob + (long)(mg + v) * GD_N + ng] = val;
                if (ti != tj) out[ob + (long)ng * GD_N + (mg + v)] = val;
            }
        }
    }
}

extern "C" void kernel_launch(void* const* d_in, const int* in_sizes, int n_in,
                              void* d_out, int out_size, void* d_ws, size_t ws_size,
                              hipStream_t stream) {
    (void)in_sizes; (void)n_in; (void)out_size; (void)ws_size;
    const float* g = (const float*)d_in[0];   // [8,8,1024,16,16] fp32
    float* out   = (float*)d_out;             // [8,1024,1024] fp32
    float* sqArr = (float*)d_ws;              // [B*C*N] = 65536 floats
    float* nrArr = sqArr + GD_B * GD_C * GD_N;

    gram_rownorm_kernel<<<(GD_B * GD_C * GD_N) / 32, 256, 0, stream>>>(g, sqArr, nrArr);

    dim3 grid(36 /* upper-tri 128x128 tile pairs */, 1, GD_B);
    gram_gemm_kernel<<<grid, 256, 0, stream>>>(g, sqArr, nrArr, out);
}
